// ELM_AE_FatSpectral_Ensemble_18184891531774
// MI455X (gfx1250) — matrix operand order repaired and checked by reading, not compiled, not run on hardware
//
#include <hip/hip_runtime.h>
#include <hip/hip_bf16.h>
#include <math.h>

// CDNA5 / gfx1250: wave32, WMMA 16x16x32 f16 -> f32, TDM tensor_load_to_lds.
typedef __attribute__((ext_vector_type(16))) _Float16 v16h;
typedef __attribute__((ext_vector_type(8)))  float    v8f;
typedef __attribute__((ext_vector_type(4)))  unsigned int u32x4;
typedef __attribute__((ext_vector_type(4)))  int      i32x4;
typedef __attribute__((ext_vector_type(8)))  int      i32x8;

#define BATCH 32
#define SP    14
#define WHN   196      // 14*14
#define WHP   224      // padded to multiple of 32 (K granularity)
#define QH    16       // ELM hidden size
#define OUTC  3840     // 256+512+1024+2048

// ---------------------------------------------------------------------------
// TDM: stage a 2D f16 tile (rows x rowlen, global row stride = srcstride
// elements) into LDS, contiguous row-major. Descriptor per ISA 8.3/8.4.
// Falls back to a cooperative copy when the builtin is unavailable.
// ---------------------------------------------------------------------------
__device__ __forceinline__ void tdm_stage_f16(_Float16* lds, const _Float16* g,
                                              unsigned rows, unsigned rowlen,
                                              unsigned srcstride, int lane) {
#if __has_builtin(__builtin_amdgcn_tensor_load_to_lds)
  unsigned long long ga = (unsigned long long)(uintptr_t)g;
  unsigned lds_off = (unsigned)(uintptr_t)lds;   // low 32 bits = LDS byte offset
  u32x4 g0;
  g0[0] = 1u;                                    // count=1, user mode, no gather
  g0[1] = lds_off;                               // lds_addr (bytes)
  g0[2] = (unsigned)(ga & 0xffffffffu);          // global_addr[31:0]
  g0[3] = (unsigned)((ga >> 32) & 0x01ffffffu)   // global_addr[56:32]
        | (2u << 30);                            // type = 2 ("image")
  i32x8 g1;
  g1[0] = (int)(1u << 16);                       // data_size=1 (2 bytes)
  g1[1] = (int)((rowlen & 0xffffu) << 16);       // tensor_dim0[15:0]
  g1[2] = (int)(((rowlen >> 16) & 0xffffu) |     // tensor_dim0[31:16]
                ((rows & 0xffffu) << 16));       // tensor_dim1[15:0]
  g1[3] = (int)(((rows >> 16) & 0xffffu) |       // tensor_dim1[31:16]
                ((rowlen & 0xffffu) << 16));     // tile_dim0
  g1[4] = (int)(rows & 0xffffu);                 // tile_dim1 (tile_dim2=0)
  g1[5] = (int)srcstride;                        // tensor_dim0_stride[31:0]
  g1[6] = 0;                                     // stride0[47:32], dim1_stride lo
  g1[7] = 0;
  i32x4 z4 = {0, 0, 0, 0};
#if __clang_major__ >= 23
  i32x8 z8 = {0, 0, 0, 0, 0, 0, 0, 0};
  __builtin_amdgcn_tensor_load_to_lds(g0, g1, z4, z4, z8, 0);
#else
  __builtin_amdgcn_tensor_load_to_lds(g0, g1, z4, z4, 0);
#endif
#else
  for (unsigned i = (unsigned)lane; i < rows * rowlen; i += 32)
    lds[i] = g[(i / rowlen) * (size_t)srcstride + (i % rowlen)];
#endif
}

__device__ __forceinline__ void tdm_wait() {
#if __has_builtin(__builtin_amdgcn_s_wait_tensorcnt)
  __builtin_amdgcn_s_wait_tensorcnt(0);
#endif
  __syncthreads();
}

// ---------------------------------------------------------------------------
// Bilinear tap (half-pixel centers, clamped), computed on the fly.
// ---------------------------------------------------------------------------
__device__ __forceinline__ float bilin_tap(const float* __restrict__ x,
                                           int b, int ch, int C, int H, int W,
                                           int oy, int ox) {
  float sy = (float)H / (float)SP;
  float sx = (float)W / (float)SP;
  float fy = ((float)oy + 0.5f) * sy - 0.5f;
  float fx = ((float)ox + 0.5f) * sx - 0.5f;
  int y0 = (int)floorf(fy); float wy = fy - (float)y0;
  int x0 = (int)floorf(fx); float wx = fx - (float)x0;
  int y1 = y0 + 1, x1 = x0 + 1;
  y0 = min(max(y0, 0), H - 1); y1 = min(max(y1, 0), H - 1);
  x0 = min(max(x0, 0), W - 1); x1 = min(max(x1, 0), W - 1);
  const float* p = x + ((size_t)b * C + ch) * (size_t)(H * W);
  float v00 = p[y0 * W + x0], v01 = p[y0 * W + x1];
  float v10 = p[y1 * W + x0], v11 = p[y1 * W + x1];
  return (1.0f - wy) * ((1.0f - wx) * v00 + wx * v01) +
         wy         * ((1.0f - wx) * v10 + wx * v11);
}

// ---------------------------------------------------------------------------
// 1) Per-(b,pixel) mean & 1/std over channels (ddof=1), fused with resize.
// ---------------------------------------------------------------------------
__global__ void stats_kernel(const float* __restrict__ x, int C, int H, int W,
                             float* __restrict__ mu, float* __restrict__ inv) {
  int id = blockIdx.x * blockDim.x + threadIdx.x;
  if (id >= BATCH * WHN) return;
  int b = id / WHN, pix = id % WHN;
  int oy = pix / SP, ox = pix % SP;
  float s1 = 0.0f, s2 = 0.0f;
  for (int ch = 0; ch < C; ++ch) {
    // Hint the streaming channel-major walk (global_prefetch_b8 on gfx1250).
    __builtin_prefetch(x + ((size_t)b * C + ch + 4) * (size_t)(H * W), 0, 0);
    float v = bilin_tap(x, b, ch, C, H, W, oy, ox);
    s1 += v; s2 += v * v;
  }
  float m   = s1 / (float)C;
  float var = (s2 - (float)C * m * m) / (float)(C - 1);
  float iv  = 1.0f / sqrtf(var);
  if (!__builtin_isfinite(iv)) iv = 0.0f;   // nan_to_num surrogate
  mu[id]  = m;
  inv[id] = iv;
}

// ---------------------------------------------------------------------------
// 2) Z-score normalize -> f16, padded [b, C, WHP] with zero pad columns.
// ---------------------------------------------------------------------------
__global__ void norm_kernel(const float* __restrict__ x, int C, int H, int W,
                            const float* __restrict__ mu,
                            const float* __restrict__ inv,
                            _Float16* __restrict__ xn) {
  int id = blockIdx.x * blockDim.x + threadIdx.x;
  int total = BATCH * C * WHP;
  if (id >= total) return;
  int col = id % WHP;
  int ch  = (id / WHP) % C;
  int b   = id / (WHP * C);
  _Float16 o = (_Float16)0.0f;
  if (col < WHN) {
    float v = bilin_tap(x, b, ch, C, H, W, col / SP, col % SP);
    float z = (v - mu[b * WHN + col]) * inv[b * WHN + col];
    if (!__builtin_isfinite(z)) z = 0.0f;
    o = (_Float16)z;
  }
  xn[id] = o;
}

// ---------------------------------------------------------------------------
// Fragment index helpers per ISA 7.12.2 (16-bit A 16x32, 16-bit B 32x16).
// half index i -> K offset within the 32-wide K slice.
// ---------------------------------------------------------------------------
__device__ __forceinline__ int frag_a_k(int i, int hi) {
  int v = i >> 1, j = i & 1;
  return (v < 4) ? (hi * 8 + v * 2 + j) : (16 + hi * 8 + (v - 4) * 2 + j);
}
__device__ __forceinline__ int frag_b_k(int i, int hi) {
  int v = i >> 1, j = i & 1;
  return (v < 4) ? (hi * 8 + v * 2 + j) : (16 + hi * 8 + (v - 4) * 2 + j);
}

// ---------------------------------------------------------------------------
// 3) H = sigmoid(W @ Xn): one wave per (b, 16-col pixel tile). TDM stages the
//    strided [C,16] Xn column tile into LDS; K-loop over C with WMMA.
// ---------------------------------------------------------------------------
__global__ __launch_bounds__(32)
void h_gemm_kernel(const float* __restrict__ Wm, const _Float16* __restrict__ xn,
                   int C, _Float16* __restrict__ Hout) {
  extern __shared__ _Float16 xT[];             // C*16 halves
  int lane = threadIdx.x, hi = lane >> 4, mn = lane & 15;
  int ntile = blockIdx.x;            // 0..13
  int b     = blockIdx.y;            // 0..31
  const _Float16* X = xn + (size_t)b * C * WHP + ntile * 16;
  tdm_stage_f16(xT, X, (unsigned)C, 16u, WHP, lane);   // async DMA -> LDS
  tdm_wait();
  v8f acc = {};
  for (int k0 = 0; k0 < C; k0 += 32) {
    v16h a, bb;
#pragma unroll
    for (int i = 0; i < 16; ++i) {
      a[i]  = (_Float16)Wm[mn * C + k0 + frag_a_k(i, hi)];
      bb[i] = xT[(k0 + frag_b_k(i, hi)) * 16 + mn];
    }
    acc = __builtin_amdgcn_wmma_f32_16x16x32_f16(false, a, false, bb,
                                                 (short)0, acc, false, false);
  }
#pragma unroll
  for (int r = 0; r < 8; ++r) {
    int m = r + 8 * hi;
    int n = ntile * 16 + mn;
    float hv = 1.0f / (1.0f + __expf(-acc[r]));
    Hout[(size_t)b * QH * WHP + m * WHP + n] =
        (n < WHN) ? (_Float16)hv : (_Float16)0.0f;
  }
}

// ---------------------------------------------------------------------------
// 4) G = H H^T via WMMA (H staged by TDM), then 16x16 Gauss-Jordan inversion
//    in LDS (one wave per sample). pinv(H) = H^T G^-1 for full-row-rank H.
// ---------------------------------------------------------------------------
__global__ __launch_bounds__(32)
void gram_inv_kernel(const _Float16* __restrict__ Hm, float* __restrict__ Ginv) {
  __shared__ _Float16 hT[QH * WHP];            // 7 KB
  __shared__ float aug[16][33];
  int lane = threadIdx.x, hi = lane >> 4, nn = lane & 15;
  int b = blockIdx.x;
  tdm_stage_f16(hT, Hm + (size_t)b * QH * WHP, QH, WHP, WHP, lane);
  tdm_wait();
  v8f acc = {};
  for (int k0 = 0; k0 < WHP; k0 += 32) {
    v16h a, bb;
#pragma unroll
    for (int i = 0; i < 16; ++i) {
      a[i]  = hT[nn * WHP + k0 + frag_a_k(i, hi)];   // A = H[m][k], m = nn
      bb[i] = hT[nn * WHP + k0 + frag_b_k(i, hi)];   // B = H^T[k][n] = H[n][k]
    }
    acc = __builtin_amdgcn_wmma_f32_16x16x32_f16(false, a, false, bb,
                                                 (short)0, acc, false, false);
  }
#pragma unroll
  for (int r = 0; r < 8; ++r) {
    int m = r + 8 * hi;
    aug[m][nn]      = acc[r];
    aug[m][16 + nn] = (m == nn) ? 1.0f : 0.0f;
  }
  __syncthreads();
  for (int k = 0; k < 16; ++k) {
    float piv = aug[k][k];
    float pin = 1.0f / (piv + 1e-20f);
    __syncthreads();
    aug[k][lane] *= pin;
    __syncthreads();
    for (int i = 0; i < 16; ++i) {
      if (i != k) {
        float f = aug[i][k];              // broadcast read precedes writes
        aug[i][lane] -= f * aug[k][lane];
      }
    }
    __syncthreads();
  }
#pragma unroll
  for (int r = 0; r < 8; ++r) {
    int m = r + 8 * hi;
    Ginv[(size_t)b * 256 + m * 16 + nn] = aug[m][16 + nn];
  }
}

// ---------------------------------------------------------------------------
// 5) T = Xn H^T (both tiles TDM-staged), then Bmat = T G^-1 (WMMA via LDS
//    relayout, K padded 16->32), then variance over 16 AE columns (ddof=1).
// ---------------------------------------------------------------------------
__global__ __launch_bounds__(32)
void tvar_kernel(const _Float16* __restrict__ xn, const _Float16* __restrict__ Hm,
                 const float* __restrict__ Ginv, int C,
                 float* __restrict__ out, int out_off) {
  __shared__ _Float16 xT[16 * WHP];            // 7 KB: this tile's 16 channels
  __shared__ _Float16 hT[QH * WHP];            // 7 KB
  __shared__ float tls[16][17];
  int lane = threadIdx.x, hi = lane >> 4, nn = lane & 15;
  int mtile = blockIdx.x;            // 0..C/16-1
  int b     = blockIdx.y;
  const _Float16* X = xn + (size_t)b * C * WHP + (size_t)mtile * 16 * WHP;
  tdm_stage_f16(xT, X, 16u, WHP, WHP, lane);
  tdm_stage_f16(hT, Hm + (size_t)b * QH * WHP, QH, WHP, WHP, lane);
  tdm_wait();
  v8f acc = {};
  for (int k0 = 0; k0 < WHP; k0 += 32) {
    v16h a, bb;
#pragma unroll
    for (int i = 0; i < 16; ++i) {
      a[i]  = xT[nn * WHP + k0 + frag_a_k(i, hi)];   // Xn rows (channels)
      bb[i] = hT[nn * WHP + k0 + frag_b_k(i, hi)];   // H^T[k][n]
    }
    acc = __builtin_amdgcn_wmma_f32_16x16x32_f16(false, a, false, bb,
                                                 (short)0, acc, false, false);
  }
#pragma unroll
  for (int r = 0; r < 8; ++r) tls[r + 8 * hi][nn] = acc[r];
  __syncthreads();

  // Second WMMA: (16x16) T times (16x16) G^-1, K zero-padded to 32.
  const float* G = Ginv + (size_t)b * 256;
  v16h a2, b2;
#pragma unroll
  for (int i = 0; i < 16; ++i) {
    int ka = frag_a_k(i, hi);
    a2[i] = (ka < 16) ? (_Float16)tls[nn][ka] : (_Float16)0.0f;
    int kb = frag_b_k(i, hi);
    b2[i] = (kb < 16) ? (_Float16)G[kb * 16 + nn] : (_Float16)0.0f;
  }
  __syncthreads();
  v8f d = {};
  d = __builtin_amdgcn_wmma_f32_16x16x32_f16(false, a2, false, b2,
                                             (short)0, d, false, false);
#pragma unroll
  for (int r = 0; r < 8; ++r) tls[r + 8 * hi][nn] = d[r];
  __syncthreads();

  if (lane < 16) {
    float s = 0.0f;
#pragma unroll
    for (int j = 0; j < 16; ++j) s += tls[lane][j];
    float m = s / 16.0f, ss = 0.0f;
#pragma unroll
    for (int j = 0; j < 16; ++j) { float dd = tls[lane][j] - m; ss += dd * dd; }
    out[(size_t)b * OUTC + out_off + mtile * 16 + lane] = ss / 15.0f;
  }
}

// ---------------------------------------------------------------------------
// Host launcher: scales processed sequentially on `stream`; workspace reused.
// d_in order from setup_inputs(): x0,W0,x1,W1,x2,W2,x3,W3.
// ---------------------------------------------------------------------------
extern "C" void kernel_launch(void* const* d_in, const int* in_sizes, int n_in,
                              void* d_out, int out_size, void* d_ws, size_t ws_size,
                              hipStream_t stream) {
  (void)in_sizes; (void)n_in; (void)out_size; (void)ws_size;
  static const int Cs[4]   = {256, 512, 1024, 2048};
  static const int Hs[4]   = {56, 28, 14, 7};
  static const int offs[4] = {0, 256, 768, 1792};

  char* ws = (char*)d_ws;
  _Float16* xn = (_Float16*)ws;
  size_t o = (size_t)BATCH * 2048 * WHP * sizeof(_Float16);   // 29.4 MB (max C)
  float* mu  = (float*)(ws + o); o += (size_t)BATCH * WHN * sizeof(float);
  float* inv = (float*)(ws + o); o += (size_t)BATCH * WHN * sizeof(float);
  _Float16* Hbuf = (_Float16*)(ws + o);
  o += (size_t)BATCH * QH * WHP * sizeof(_Float16);
  float* Ginv = (float*)(ws + o);

  float* out = (float*)d_out;

  for (int m = 0; m < 4; ++m) {
    const float* x  = (const float*)d_in[2 * m];
    const float* Wm = (const float*)d_in[2 * m + 1];
    int C = Cs[m], H = Hs[m];

    int nstat = (BATCH * WHN + 255) / 256;
    stats_kernel<<<nstat, 256, 0, stream>>>(x, C, H, H, mu, inv);

    int tot = BATCH * C * WHP;
    norm_kernel<<<(tot + 255) / 256, 256, 0, stream>>>(x, C, H, H, mu, inv, xn);

    size_t hshm = (size_t)C * 16 * sizeof(_Float16);          // <= 64 KB
    h_gemm_kernel<<<dim3(WHP / 16, BATCH), 32, hshm, stream>>>(Wm, xn, C, Hbuf);

    gram_inv_kernel<<<BATCH, 32, 0, stream>>>(Hbuf, Ginv);

    tvar_kernel<<<dim3(C / 16, BATCH), 32, 0, stream>>>(xn, Hbuf, Ginv, C,
                                                        out, offs[m]);
  }
}
